// PathRestore_43619687858891
// MI455X (gfx1250) — matrix operand order repaired and checked by reading, not compiled, not run on hardware
//
#include <hip/hip_runtime.h>

typedef __attribute__((ext_vector_type(16))) _Float16 v16h;
typedef __attribute__((ext_vector_type(8)))  _Float16 v8h;
typedef __attribute__((ext_vector_type(8)))  float    v8f;
typedef __attribute__((ext_vector_type(4)))  unsigned int v4u;
typedef __attribute__((ext_vector_type(8)))  int      v8i;
typedef __attribute__((ext_vector_type(4)))  int      v4i;

#define NB   16          // batch
#define HH   128
#define WW   128
#define CC   64
#define NPIX (NB*HH*WW)  // 262144
#define NEL  (NPIX*CC)   // 16777216 (2^24)
#define HP   130         // padded spatial extent (1-px halo)
#define NPAD ((size_t)NB*HP*HP*CC)
#define EPSF 1e-5f

// padded NHWC f16 offset for interior pixel (pix, c)
__device__ __forceinline__ size_t padOff(int pix, int c)
{
    int b = pix >> 14, rem = pix & 16383, yy = rem >> 7, xx = rem & 127;
    return ((size_t)((b * HP + yy + 1) * HP) + xx + 1) * 64 + c;
}

// ---------------------------------------------------------------------------
// WMMA implicit-GEMM 3x3 conv, 64->64, SAME. Halo-padded NHWC f16 in ->
// dense NHWC f32 out. grid = NB*HH*2 (b, row, cout-half); 8 waves = 8 x-tiles.
// Each wave register-blocks 2 cout tiles (32 couts) -> 36 WMMAs, A reused 2x.
// Weight tile (32x576 f16 = 36 KB) is DMA'd into LDS by the Tensor Data
// Mover (1-D descriptor, 9216 dwords), completed with s_wait_tensorcnt.
// Fuses per-channel sum/sum^2 for BN (batch or per-sample).
// ---------------------------------------------------------------------------
__launch_bounds__(256)
__global__ void conv64_wmma_kernel(const _Float16* __restrict__ act,   // padded
                                   const _Float16* __restrict__ wpack, // [64][576] f16
                                   float* __restrict__ out,            // dense NHWC f32
                                   float* __restrict__ acc,            // [128] or [16][128]
                                   int perSample)
{
    __shared__ _Float16 ldsW[32 * 576];   // 36 KB: [n_local][k]
    __shared__ float ldsS[32], ldsS2[32];

    const int tid  = threadIdx.x;
    const int lane = tid & 31;
    const int wv   = tid >> 5;            // x-tile 0..7
    const int gid  = blockIdx.x;
    const int nT2  = gid & 1;             // cout half (0: 0-31, 1: 32-63)
    const int yy   = (gid >> 1) & 127;
    const int b    = gid >> 8;

    if (tid < 32) { ldsS[tid] = 0.f; ldsS2[tid] = 0.f; }

    // ---- TDM: DMA the 36 KB weight tile into LDS (issued by wave 0) ----
    {
        unsigned long long ga = (unsigned long long)(const void*)
                                (wpack + (size_t)nT2 * (32 * 576));
        unsigned ldsAddr = (unsigned)(size_t)(void*)ldsW;
        v4u g0;
        g0[0] = 1u;                              // count=1, user descriptor
        g0[1] = ldsAddr;                         // lds_addr
        g0[2] = (unsigned)ga;                    // global_addr[31:0]
        g0[3] = (unsigned)(ga >> 32) | (2u << 30); // global_addr[56:32] | type=2
        v8i g1;
        g1[0] = 0x20000;                         // data_size=2 (4-byte units)
        g1[1] = (9216 & 0xFFFF) << 16;           // tensor_dim0 = 9216 dwords
        g1[2] = (1 << 16);                       // tensor_dim1 = 1
        g1[3] = (9216 << 16);                    // tile_dim0 = 9216
        g1[4] = 1 | (1 << 16);                   // tile_dim1 = 1, tile_dim2 = 1
        g1[5] = 9216;                            // tensor_dim0_stride
        g1[6] = 0;
        g1[7] = 0;
        v4i z4 = {0, 0, 0, 0};
        v8i z8 = {0, 0, 0, 0, 0, 0, 0, 0};
        if (tid < 32) {
            __builtin_amdgcn_tensor_load_to_lds(g0, g1, z4, z4, z8, 0);
            __builtin_amdgcn_s_wait_tensorcnt(0);
        }
    }
    __syncthreads();

    const int m    = lane & 15;   // pixel-in-tile == cout-in-tile
    const int half = lane >> 4;
    const int x0   = wv * 16;

    v8f c0 = {}, c1 = {};
    for (int p = 0; p < 9; ++p) {
        const int ky = p / 3;     // padded row = yy + ky
        const int kx = p % 3;     // padded col = x0 + m + kx
        const _Float16* ap = act +
            (((size_t)(b * HP + yy + ky) * HP) + (x0 + m + kx)) * 64 + half * 16;
        #pragma unroll
        for (int h32 = 0; h32 < 2; ++h32) {
            // A k order chosen here (B matches): lane half -> 16 contiguous K {0..15} / {16..31}
            union { v16h v; v8h h[2]; } a;
            a.h[0] = *(const v8h*)(ap + h32 * 32);
            a.h[1] = *(const v8h*)(ap + h32 * 32 + 8);
            const int k0 = p * 64 + h32 * 32;
            // B: lane n = lane&15, half -> 16 contiguous K, same k order as A
            v16h b0 = *(const v16h*)(ldsW + m * 576 + k0 + half * 16);
            v16h b1 = *(const v16h*)(ldsW + (16 + m) * 576 + k0 + half * 16);
            c0 = __builtin_amdgcn_wmma_f32_16x16x32_f16(false, a.v, false, b0,
                                                        (short)0, c0, false, false);
            c1 = __builtin_amdgcn_wmma_f32_16x16x32_f16(false, a.v, false, b1,
                                                        (short)0, c1, false, false);
        }
    }

    // D layout: lane holds M = j + 8*half (pixel), N = lane&15 (cout)
    float* op = out + (((size_t)(b * 128 + yy) * 128 + x0) * 64) + nT2 * 32 + m;
    float s0 = 0.f, q0 = 0.f, s1 = 0.f, q1 = 0.f;
    #pragma unroll
    for (int j = 0; j < 8; ++j) {
        float v0 = c0[j], v1 = c1[j];
        s0 += v0; q0 += v0 * v0;
        s1 += v1; q1 += v1 * v1;
        op[(size_t)(j + 8 * half) * 64]      = v0;
        op[(size_t)(j + 8 * half) * 64 + 16] = v1;
    }
    atomicAdd(&ldsS[m], s0);       atomicAdd(&ldsS2[m], q0);
    atomicAdd(&ldsS[16 + m], s1);  atomicAdd(&ldsS2[16 + m], q1);
    __syncthreads();
    if (tid < 32) {
        float* dst = perSample ? (acc + b * 128) : acc;
        atomicAdd(&dst[nT2 * 32 + tid], ldsS[tid]);
        atomicAdd(&dst[64 + nT2 * 32 + tid], ldsS2[tid]);
    }
}

// --------------------------- BN helpers ------------------------------------
__device__ __forceinline__ float2 bn_coef(const float* acc, const float* g,
                                          const float* be, int c, float invCount)
{
    float mean = acc[c] * invCount;
    float var  = acc[64 + c] * invCount - mean * mean;
    float sc   = g[c] * rsqrtf(var + EPSF);
    return make_float2(sc, be[c] - mean * sc);
}

// y -> relu(scale*y + shift [+ addin]); writes padded f16 (+optional dense f32)
__launch_bounds__(256)
__global__ void bn_apply_kernel(const float* __restrict__ y, const float* __restrict__ acc,
                                const float* __restrict__ g, const float* __restrict__ be,
                                const float* __restrict__ addin, float* __restrict__ out32,
                                _Float16* __restrict__ out16, float invCount)
{
    int idx = blockIdx.x * 256 + threadIdx.x;
    int c = idx & 63;
    float2 k = bn_coef(acc, g, be, c, invCount);
    float v = k.x * y[idx] + k.y;
    if (addin) v += addin[idx];
    v = fmaxf(v, 0.f);
    if (out32) out32[idx] = v;
    out16[padOff(idx >> 6, c)] = (_Float16)v;
}

// per-sample BN (stats over H,W per sample) -> relu -> padded f16
__launch_bounds__(256)
__global__ void bn_apply_ps_kernel(const float* __restrict__ y, const float* __restrict__ accPS,
                                   const float* __restrict__ g, const float* __restrict__ be,
                                   _Float16* __restrict__ out16)
{
    int idx = blockIdx.x * 256 + threadIdx.x;
    int c = idx & 63;
    int b = idx >> 20;                      // 2^20 elements per sample
    float2 k = bn_coef(accPS + b * 128, g, be, c, 1.f / 16384.f);
    out16[padOff(idx >> 6, c)] = (_Float16)fmaxf(k.x * y[idx] + k.y, 0.f);
}

// expert = relu(psBN(e2) + shared); chosen = action ? expert : shared;
// fuses batch sum/sum^2 of chosen for the block-output BN.
__launch_bounds__(256)
__global__ void select_kernel(const float* __restrict__ e2, const float* __restrict__ accPS,
                              const float* __restrict__ g, const float* __restrict__ be,
                              const float* __restrict__ shared32, const int* __restrict__ action,
                              float* __restrict__ chosen, float* __restrict__ accB)
{
    __shared__ float ldsS[64], ldsS2[64];
    int t = threadIdx.x;
    if (t < 64) { ldsS[t] = 0.f; ldsS2[t] = 0.f; }
    __syncthreads();
    int c = t & 63;
    float s = 0.f, s2 = 0.f;
    int base = blockIdx.x * 4096;
    for (int k = 0; k < 16; ++k) {
        int idx = base + k * 256 + t;       // c == t&63 invariant
        int b = idx >> 20;
        float2 kc = bn_coef(accPS + b * 128, g, be, c, 1.f / 16384.f);
        float sv = shared32[idx];
        float ev = fmaxf(kc.x * e2[idx] + kc.y + sv, 0.f);
        float v  = (action[b] != 0) ? ev : sv;
        chosen[idx] = v;
        s += v; s2 += v * v;
    }
    atomicAdd(&ldsS[c], s); atomicAdd(&ldsS2[c], s2);
    __syncthreads();
    if (t < 64) { atomicAdd(&accB[t], ldsS[t]); atomicAdd(&accB[64 + t], ldsS2[t]); }
}

// conv1: 3->64, NCHW f32 in -> dense NHWC f32 out, fused batch stats
__launch_bounds__(256)
__global__ void conv1_kernel(const float* __restrict__ x, const float* __restrict__ w,
                             float* __restrict__ y, float* __restrict__ accB)
{
    __shared__ float ldsS[64], ldsS2[64];
    int t = threadIdx.x;
    if (t < 64) { ldsS[t] = 0.f; ldsS2[t] = 0.f; }
    __syncthreads();
    int c = t & 63;
    float wr[27];
    #pragma unroll
    for (int i = 0; i < 27; ++i) wr[i] = w[c * 27 + i];
    float s = 0.f, s2 = 0.f;
    int base = blockIdx.x * 4096;
    for (int k = 0; k < 16; ++k) {
        int idx = base + k * 256 + t;
        int pix = idx >> 6;
        int b = pix >> 14, rem = pix & 16383, yy = rem >> 7, xx = rem & 127;
        float v = 0.f;
        for (int ci = 0; ci < 3; ++ci)
            for (int ky = 0; ky < 3; ++ky) {
                int ys = yy + ky - 1; if ((unsigned)ys >= 128u) continue;
                for (int kx = 0; kx < 3; ++kx) {
                    int xs = xx + kx - 1; if ((unsigned)xs >= 128u) continue;
                    v += x[((b * 3 + ci) * 128 + ys) * 128 + xs] * wr[ci * 9 + ky * 3 + kx];
                }
            }
        y[idx] = v; s += v; s2 += v * v;
    }
    atomicAdd(&ldsS[c], s); atomicAdd(&ldsS2[c], s2);
    __syncthreads();
    if (t < 64) { atomicAdd(&accB[t], ldsS[t]); atomicAdd(&accB[64 + t], ldsS2[t]); }
}

// conv2: 64->3, dense NHWC f32 in -> NCHW f32 out (with bias; no BN after)
__launch_bounds__(256)
__global__ void conv2_kernel(const float* __restrict__ h, const float* __restrict__ w,
                             const float* __restrict__ bias, float* __restrict__ out)
{
    int idx = blockIdx.x * 256 + threadIdx.x;      // 786432
    int b = idx / 49152, r = idx % 49152;
    int co = r >> 14, yy = (r >> 7) & 127, xx = r & 127;
    float s = bias[co];
    for (int ky = 0; ky < 3; ++ky) {
        int ys = yy + ky - 1; if ((unsigned)ys >= 128u) continue;
        for (int kx = 0; kx < 3; ++kx) {
            int xs = xx + kx - 1; if ((unsigned)xs >= 128u) continue;
            const float* hp = h + ((size_t)((b * 128 + ys) * 128 + xs)) * 64;
            const float* wp = w + co * 576 + ky * 3 + kx;   // ci stride 9
            float a = 0.f;
            for (int ci = 0; ci < 64; ++ci) a += hp[ci] * wp[ci * 9];
            s += a;
        }
    }
    out[idx] = s;
}

// OIHW f32 [64][64][3][3] -> f16 [co][k], k = (ky*3+kx)*64 + ci
__launch_bounds__(256)
__global__ void pack_w_kernel(const float* __restrict__ w, _Float16* __restrict__ wp)
{
    int idx = blockIdx.x * 256 + threadIdx.x;      // 36864
    int co = idx / 576, k = idx % 576;
    int p = k >> 6, ci = k & 63;
    wp[idx] = (_Float16)w[co * 576 + ci * 9 + p];
}

// ------------------------ path-finder (tiny) --------------------------------
// reads the zero halo instead of bounds-checking
__launch_bounds__(256)
__global__ void pf_conv1_kernel(const _Float16* __restrict__ h16, const float* __restrict__ w,
                                float* __restrict__ out)
{
    int idx = blockIdx.x * 256 + threadIdx.x;      // 524288
    int co = idx & 1, pix = idx >> 1;
    int b = pix >> 14, rem = pix & 16383, yy = rem >> 7, xx = rem & 127;
    float s = 0.f;
    for (int ky = 0; ky < 3; ++ky)
        for (int kx = 0; kx < 3; ++kx) {
            const _Float16* ap = h16 +
                (((size_t)(b * HP + yy + ky) * HP) + xx + kx) * 64;
            const float* wp = w + co * 576 + ky * 3 + kx;
            float a = 0.f;
            for (int ci = 0; ci < 64; ++ci) a += (float)ap[ci] * wp[ci * 9];
            s += a;
        }
    out[idx] = s;
}

__launch_bounds__(256)
__global__ void pf_conv2_kernel(const float* __restrict__ in, const float* __restrict__ w,
                                float* __restrict__ out)
{
    int idx = blockIdx.x * 256 + threadIdx.x;      // 524288
    int co = idx & 1, pix = idx >> 1;
    int b = pix >> 14, rem = pix & 16383, yy = rem >> 7, xx = rem & 127;
    float s = 0.f;
    for (int ky = 0; ky < 3; ++ky) {
        int ys = yy + ky - 1; if ((unsigned)ys >= 128u) continue;
        for (int kx = 0; kx < 3; ++kx) {
            int xs = xx + kx - 1; if ((unsigned)xs >= 128u) continue;
            const float* ip = in + ((size_t)((b * 128 + ys) * 128 + xs)) * 2;
            for (int ci = 0; ci < 2; ++ci)
                s += ip[ci] * w[co * 18 + ci * 9 + ky * 3 + kx];
        }
    }
    out[idx] = s;
}

__launch_bounds__(256)
__global__ void stats_c2_kernel(const float* __restrict__ buf, float* __restrict__ acc4)
{
    __shared__ float l[4];
    int t = threadIdx.x;
    if (t < 4) l[t] = 0.f;
    __syncthreads();
    int c = t & 1;
    float s = 0.f, s2 = 0.f;
    int base = blockIdx.x * 8192;
    for (int k = 0; k < 32; ++k) {
        float v = buf[base + k * 256 + t];
        s += v; s2 += v * v;
    }
    atomicAdd(&l[c], s); atomicAdd(&l[2 + c], s2);
    __syncthreads();
    if (t < 4) atomicAdd(&acc4[t], l[t]);
}

__launch_bounds__(256)
__global__ void bn_relu_c2_kernel(const float* __restrict__ in, const float* __restrict__ acc4,
                                  const float* __restrict__ g, const float* __restrict__ be,
                                  float* __restrict__ out)
{
    int idx = blockIdx.x * 256 + threadIdx.x;
    int c = idx & 1;
    const float inv = 1.f / (float)NPIX;
    float mean = acc4[c] * inv;
    float var  = acc4[2 + c] * inv - mean * mean;
    float sc = g[c] * rsqrtf(var + EPSF);
    out[idx] = fmaxf(sc * (in[idx] - mean) + be[c], 0.f);
}

__launch_bounds__(256)
__global__ void router_kernel(const float* __restrict__ pf, const float* __restrict__ fw1,
                              const float* __restrict__ fb1, const float* __restrict__ fw2,
                              const float* __restrict__ fb2, int* __restrict__ action)
{
    __shared__ float gap[32];
    int t = threadIdx.x;
    if (t < 32) gap[t] = 0.f;
    __syncthreads();
    int pair = t >> 3, sub = t & 7;
    int b = pair >> 1, c = pair & 1;
    float s = 0.f;
    for (int pix = sub; pix < 16384; pix += 8)
        s += pf[((size_t)b * 16384 + pix) * 2 + c];
    atomicAdd(&gap[pair], s);
    __syncthreads();
    if (t < 16) {
        float g0 = gap[t * 2 + 0] * (1.f / 16384.f);
        float g1 = gap[t * 2 + 1] * (1.f / 16384.f);
        float l0 = fb2[0], l1 = fb2[1];
        for (int j = 0; j < 64; ++j) {
            float hj = fw1[j * 2 + 0] * g0 + fw1[j * 2 + 1] * g1 + fb1[j];
            l0 += fw2[j] * hj;
            l1 += fw2[64 + j] * hj;
        }
        action[t] = (l1 > l0) ? 1 : 0;   // argmax, first index wins ties
    }
}

// ---------------------------------------------------------------------------
extern "C" void kernel_launch(void* const* d_in, const int* in_sizes, int n_in,
                              void* d_out, int out_size, void* d_ws, size_t ws_size,
                              hipStream_t stream)
{
    (void)in_sizes; (void)n_in; (void)out_size;
    const float* const* in = (const float* const*)d_in;

    // ---- parameter indices (setup_inputs dict insertion order, depth-first)
    const float* X     = in[0];
    const float* c1w   = in[1];
    const float* bn1g  = in[3];
    const float* bn1b  = in[4];
    const float* pf_w1 = in[5];
    const float* pf_g1 = in[7];
    const float* pf_be1= in[8];
    const float* pf_w2 = in[9];
    const float* pf_g2 = in[11];
    const float* pf_be2= in[12];
    const float* fw1   = in[13];
    const float* fb1   = in[14];
    const float* fw2   = in[15];
    const float* fb2   = in[16];
    const float* c2w   = in[17 + 18 * 6];
    const float* c2b   = in[17 + 18 * 6 + 1];

    // ---- workspace carve-up (~310 MB needed)
    char* ws = (char*)d_ws;
    size_t off = 0;
    auto alloc = [&](size_t bytes) { size_t o = off; off += (bytes + 255) & ~(size_t)255; return o; };
    float*     A32  = (float*)    (ws + alloc((size_t)NEL * 4));   // h master f32 / chosen
    float*     S32  = (float*)    (ws + alloc((size_t)NEL * 4));   // shared f32
    float*     Y32  = (float*)    (ws + alloc((size_t)NEL * 4));   // conv output scratch
    _Float16*  X16a = (_Float16*) (ws + alloc(NPAD * 2));          // h f16 (padded)
    _Float16*  X16b = (_Float16*) (ws + alloc(NPAD * 2));          // temp f16 (padded)
    _Float16*  X16s = (_Float16*) (ws + alloc(NPAD * 2));          // shared f16 (padded)
    float*     PFa  = (float*)    (ws + alloc((size_t)NPIX * 2 * 4));
    float*     PFb  = (float*)    (ws + alloc((size_t)NPIX * 2 * 4));
    _Float16*  WPK  = (_Float16*) (ws + alloc((size_t)64 * 576 * 2));
    float*     ACCB = (float*)    (ws + alloc(128 * 4));
    float*     ACCPS= (float*)    (ws + alloc((size_t)NB * 128 * 4));
    int*       ACT  = (int*)      (ws + alloc(NB * 4));
    if (off > ws_size) { /* insufficient scratch; nothing safe to do */ }

    const float invB  = 1.f / (float)NPIX;     // batch BN count
    const dim3 blk256(256);
    const dim3 gConv(NB * HH * 2);             // 4096 WMMA blocks (2 cout tiles/wave)
    const dim3 gElem(NEL / 256);               // 65536
    const dim3 gRed(4096);

    // zero padded f16 buffers once per launch (halo must be 0)
    (void)hipMemsetAsync(X16a, 0, NPAD * 2, stream);
    (void)hipMemsetAsync(X16b, 0, NPAD * 2, stream);
    (void)hipMemsetAsync(X16s, 0, NPAD * 2, stream);

    // ================= stem: h = relu(BN(conv1(x))) =================
    (void)hipMemsetAsync(ACCB, 0, 128 * 4, stream);
    conv1_kernel<<<dim3(NEL / 4096), blk256, 0, stream>>>(X, c1w, Y32, ACCB);
    bn_apply_kernel<<<gElem, blk256, 0, stream>>>(Y32, ACCB, bn1g, bn1b,
                                                  nullptr, A32, X16a, invB);

    // ================= 6 routed residual blocks =================
    for (int blkI = 0; blkI < 6; ++blkI) {
        const float* const* p = in + 17 + 18 * blkI;
        const float *sh_w1 = p[0],  *sh_g1 = p[2],  *sh_be1 = p[3];
        const float *sh_w2 = p[4],  *sh_g2 = p[6],  *sh_be2 = p[7];
        const float *ex_w1 = p[8],  *ex_g1 = p[10], *ex_be1 = p[11];
        const float *ex_w2 = p[12], *ex_g2 = p[14], *ex_be2 = p[15];
        const float *outg  = p[16], *outb  = p[17];

        // --- shared res-block (batch BN) ---
        (void)hipMemsetAsync(ACCB, 0, 128 * 4, stream);
        pack_w_kernel<<<dim3(144), blk256, 0, stream>>>(sh_w1, WPK);
        conv64_wmma_kernel<<<gConv, blk256, 0, stream>>>(X16a, WPK, Y32, ACCB, 0);
        bn_apply_kernel<<<gElem, blk256, 0, stream>>>(Y32, ACCB, sh_g1, sh_be1,
                                                      nullptr, nullptr, X16b, invB);
        (void)hipMemsetAsync(ACCB, 0, 128 * 4, stream);
        pack_w_kernel<<<dim3(144), blk256, 0, stream>>>(sh_w2, WPK);
        conv64_wmma_kernel<<<gConv, blk256, 0, stream>>>(X16b, WPK, Y32, ACCB, 0);
        bn_apply_kernel<<<gElem, blk256, 0, stream>>>(Y32, ACCB, sh_g2, sh_be2,
                                                      A32, S32, X16s, invB);   // +identity, relu

        // --- path finder on block input h ---
        (void)hipMemsetAsync(ACCB, 0, 4 * 4, stream);
        pf_conv1_kernel<<<dim3(2048), blk256, 0, stream>>>(X16a, pf_w1, PFa);
        stats_c2_kernel<<<dim3(64), blk256, 0, stream>>>(PFa, ACCB);
        bn_relu_c2_kernel<<<dim3(2048), blk256, 0, stream>>>(PFa, ACCB, pf_g1, pf_be1, PFb);
        (void)hipMemsetAsync(ACCB, 0, 4 * 4, stream);
        pf_conv2_kernel<<<dim3(2048), blk256, 0, stream>>>(PFb, pf_w2, PFa);
        stats_c2_kernel<<<dim3(64), blk256, 0, stream>>>(PFa, ACCB);
        bn_relu_c2_kernel<<<dim3(2048), blk256, 0, stream>>>(PFa, ACCB, pf_g2, pf_be2, PFb);
        router_kernel<<<dim3(1), blk256, 0, stream>>>(PFb, fw1, fb1, fw2, fb2, ACT);

        // --- expert res-block on shared (per-sample BN) ---
        (void)hipMemsetAsync(ACCPS, 0, (size_t)NB * 128 * 4, stream);
        pack_w_kernel<<<dim3(144), blk256, 0, stream>>>(ex_w1, WPK);
        conv64_wmma_kernel<<<gConv, blk256, 0, stream>>>(X16s, WPK, Y32, ACCPS, 1);
        bn_apply_ps_kernel<<<gElem, blk256, 0, stream>>>(Y32, ACCPS, ex_g1, ex_be1, X16b);
        (void)hipMemsetAsync(ACCPS, 0, (size_t)NB * 128 * 4, stream);
        pack_w_kernel<<<dim3(144), blk256, 0, stream>>>(ex_w2, WPK);
        conv64_wmma_kernel<<<gConv, blk256, 0, stream>>>(X16b, WPK, Y32, ACCPS, 1);

        // --- route, then block-output BN+relu ---
        (void)hipMemsetAsync(ACCB, 0, 128 * 4, stream);
        select_kernel<<<gRed, blk256, 0, stream>>>(Y32, ACCPS, ex_g2, ex_be2,
                                                   S32, ACT, A32, ACCB);
        bn_apply_kernel<<<gElem, blk256, 0, stream>>>(A32, ACCB, outg, outb,
                                                      nullptr, A32, X16a, invB);
    }

    // ================= head: out = conv2(h) + b =================
    conv2_kernel<<<dim3(786432 / 256), blk256, 0, stream>>>(A32, c2w, c2b, (float*)d_out);
}
